// CrissCrossAttention_1_48593259987102
// MI455X (gfx1250) — matrix operand order repaired and checked by reading, not compiled, not run on hardware
//
#include <hip/hip_runtime.h>
#include <hip/hip_bf16.h>
#include <hip/hip_fp16.h>

// Criss-Cross (axial) attention for MI455X / gfx1250, wave32 + WMMA.
// B=8, C=256, CQ=32, H=W=96. All GEMM passes use v_wmma_f32_16x16x32_f16.
// LDS tiles staged pre-swizzled into WMMA lane layout; fragment load =
// 2x ds_load_b128. Staging uses packed ds_store_b32/b64/b128 where the
// k-dimension is contiguous. C-tile stores use one base + literal offsets;
// the projection epilogue transposes through LDS for coalesced b128 stores.

#define BB 8
#define CC 256
#define CQN 32
#define HH 96
#define WW 96
#define PP (HH * WW) /* 9216 */
#define FRAG 512     /* halfs per 16x32 (A) or 32x16 (B) fragment tile */

typedef __attribute__((ext_vector_type(16))) _Float16 v16h;
typedef __attribute__((ext_vector_type(8)))  _Float16 v8h;
typedef __attribute__((ext_vector_type(4)))  _Float16 v4h;
typedef __attribute__((ext_vector_type(2)))  _Float16 v2h;
typedef __attribute__((ext_vector_type(8)))  float    v8f;

// ---- WMMA fragment layout (CDNA5 ISA §7.12.2) ----
// A 16x32 f16: lane<16 holds row M=lane, K 0..7 & 16..23; lane>=16 K 8..15 & 24..31
// B 32x16 f16: lane&15 = col N; lanes 0..15 hold K 0..7 & 16..23, lanes 16..31 K 8..15 & 24..31
// C 16x16 f32: VGPR r -> (M = (lane>=16?8:0)+r, N = lane&15)

__device__ __forceinline__ void stage_A(_Float16* frag, int m, int k, _Float16 v) {
  frag[((m & 15) | ((k & 8) << 1)) * 16 + ((k & 7) | ((k & 16) >> 1))] = v;
}
// k0 4-aligned: lane fixed, idx contiguous -> ds_store_b64
__device__ __forceinline__ void stage_A4(_Float16* frag, int m, int k0, v4h v) {
  *(v4h*)(frag + ((m & 15) | ((k0 & 8) << 1)) * 16 + ((k0 & 7) | ((k0 & 16) >> 1))) = v;
}
// k0 8-aligned: -> ds_store_b128
__device__ __forceinline__ void stage_A8(_Float16* frag, int m, int k0, v8h v) {
  *(v8h*)(frag + ((m & 15) | ((k0 & 8) << 1)) * 16 + ((k0 & 16) >> 1)) = v;
}
__device__ __forceinline__ void stage_B(_Float16* frag, int k, int n, _Float16 v) {
  frag[((n & 15) | ((k & 8) << 1)) * 16 + ((k & 7) | ((k & 16) >> 1))] = v;
}
// k0 2-aligned pair -> ds_store_b32
__device__ __forceinline__ void stage_B2(_Float16* frag, int k0, int n, v2h v) {
  *(v2h*)(frag + ((n & 15) | ((k0 & 8) << 1)) * 16 + ((k0 & 7) | ((k0 & 16) >> 1))) = v;
}
// k0 4-aligned quad -> ds_store_b64
__device__ __forceinline__ void stage_B4(_Float16* frag, int k0, int n, v4h v) {
  *(v4h*)(frag + ((n & 15) | ((k0 & 8) << 1)) * 16 + ((k0 & 7) | ((k0 & 16) >> 1))) = v;
}
// Fragment load: per-lane 32 contiguous bytes -> 2x ds_load_b128.
__device__ __forceinline__ v16h frag_load(const _Float16* frag) {
  const v8h* p = (const v8h*)(frag + (threadIdx.x & 31) * 16);
  union { v16h v; v8h h[2]; } u;
  u.h[0] = p[0];
  u.h[1] = p[1];
  return u.v;
}

__device__ __forceinline__ v8f wmma_f16(v16h a, v16h b, v8f c) {
  return __builtin_amdgcn_wmma_f32_16x16x32_f16(false, a, false, b, (short)0, c,
                                                false, false);
}

__device__ __forceinline__ v4h cvt4(float4 f) {
  v4h r;
  r[0] = (_Float16)f.x; r[1] = (_Float16)f.y;
  r[2] = (_Float16)f.z; r[3] = (_Float16)f.w;
  return r;
}

// ---------------------------------------------------------------------------
// Pass 1: fused q/k/v projection. Y[o,p] = Wstack[o,:] . X[:,p] + bias.
// 16-row output block never straddles q/k/v boundary -> uniform selection.
// Epilogue transposes the 16x128 C block through LDS -> coalesced b128 stores.
// ---------------------------------------------------------------------------
__global__ void __launch_bounds__(256) qkv_proj_kernel(
    const float* __restrict__ x,
    const float* __restrict__ wq, const float* __restrict__ bq,
    const float* __restrict__ wk, const float* __restrict__ bk,
    const float* __restrict__ wv, const float* __restrict__ bv,
    _Float16* __restrict__ qb, _Float16* __restrict__ kb, _Float16* __restrict__ vb) {
  __shared__ __align__(16) _Float16 Aw[FRAG];       // one 16x32 weight tile
  __shared__ __align__(16) _Float16 Bx[8 * FRAG];   // eight 32x16 x tiles
  const int tid = threadIdx.x;
  const int wave = tid >> 5;
  const int b = blockIdx.z;
  const int obase = blockIdx.y * 16;   // 0..319
  const int p0 = blockIdx.x * 128;

  const float* wmat; const float* bias; _Float16* dst; int ob, chans;
  if (obase < CQN)          { wmat = wq; bias = bq; dst = qb; ob = obase;           chans = CQN; }
  else if (obase < 2 * CQN) { wmat = wk; bias = bk; dst = kb; ob = obase - CQN;     chans = CQN; }
  else                      { wmat = wv; bias = bv; dst = vb; ob = obase - 2 * CQN; chans = CC; }

  const int kkb = (tid >> 4) * 2;        // x staging: 2 kk rows x 8 j cols per thread
  const int j0  = (tid & 15) * 8;

  v8f acc = {};
  for (int ks = 0; ks < 8; ++ks) {
    const int c0 = ks * 32;
    {  // weights 16x32: one 2-aligned pair per thread (512 elems / 256 thr)
      int e = tid * 2;
      int m = e >> 5, kk = e & 31;
      v2h pk;
      pk[0] = (_Float16)wmat[(ob + m) * CC + c0 + kk];
      pk[1] = (_Float16)wmat[(ob + m) * CC + c0 + kk + 1];
      *(v2h*)(Aw + ((m & 15) | ((kk & 8) << 1)) * 16 + ((kk & 7) | ((kk & 16) >> 1))) = pk;
    }
    {  // x 32x128: rows kkb,kkb+1, cols j0..j0+7 (coalesced float4 reads)
      const float* xr0 = x + (size_t)(b * CC + c0 + kkb) * PP + p0 + j0;
      const float* xr1 = xr0 + PP;
      float4 a0 = *(const float4*)xr0, a1 = *(const float4*)(xr0 + 4);
      float4 b0 = *(const float4*)xr1, b1 = *(const float4*)(xr1 + 4);
      float r0[8] = {a0.x, a0.y, a0.z, a0.w, a1.x, a1.y, a1.z, a1.w};
      float r1[8] = {b0.x, b0.y, b0.z, b0.w, b1.x, b1.y, b1.z, b1.w};
#pragma unroll
      for (int j = 0; j < 8; ++j) {
        v2h pk; pk[0] = (_Float16)r0[j]; pk[1] = (_Float16)r1[j];
        stage_B2(Bx + ((j0 + j) >> 4) * FRAG, kkb, (j0 + j) & 15, pk);
      }
    }
    __syncthreads();
    acc = wmma_f16(frag_load(Aw), frag_load(Bx + wave * FRAG), acc);
    __syncthreads();
  }
  // Epilogue: bias + transpose through LDS (Bx is dead) -> coalesced stores.
  _Float16* Ytile = Bx;  // [16 channels][128 pixels], 4KB
  {
    const int lane = tid & 31;
    const int n = lane & 15;
    const int mb = (lane >> 4) * 8;
    const int col = wave * 16 + n;
    const float* bp = bias + ob + mb;
#pragma unroll
    for (int r = 0; r < 8; ++r)
      Ytile[(mb + r) * 128 + col] = (_Float16)(acc[r] + bp[r]);
  }
  __syncthreads();
  {
    const int row = tid >> 4;        // output channel within block
    const int c16 = (tid & 15) * 8;  // pixel offset
    v8h yv = *(const v8h*)(Ytile + row * 128 + c16);
    *(v8h*)(dst + (size_t)(b * chans + ob + row) * PP + p0 + c16) = yv;
  }
}

// ---------------------------------------------------------------------------
// Pass 2: per (b,w) column: E_h[h,g] = sum_c q[c,h,w] k[c,g,w]; mask h==g.
// ---------------------------------------------------------------------------
__global__ void __launch_bounds__(256) col_logits_kernel(
    const _Float16* __restrict__ qb, const _Float16* __restrict__ kb,
    float* __restrict__ attn) {
  __shared__ __align__(16) _Float16 Aq[6 * FRAG];  // 6 h-tiles of 16x32
  __shared__ __align__(16) _Float16 Bk[6 * FRAG];  // 6 g-tiles of 32x16
  const int tid = threadIdx.x;
  const int wave = tid >> 5;
  const int lane = tid & 31;
  const int w = blockIdx.x;
  const int b = blockIdx.y;
#pragma unroll
  for (int cch = 0; cch < 3; ++cch) {   // 3072 elems, 4-aligned k chunks
    int e = (tid * 3 + cch) * 4;
    int m = e >> 5, kk = e & 31;        // Aq[h=m][c=kk..kk+3]
    v4h pk;
#pragma unroll
    for (int j = 0; j < 4; ++j)
      pk[j] = qb[(b * CQN + kk + j) * PP + m * WW + w];
    stage_A4(Aq + (m >> 4) * FRAG, m & 15, kk, pk);
    int k2 = e / 96, g = e % 96;        // Bk[c=k2][g..g+3]: n varies -> scalar stores
#pragma unroll
    for (int j = 0; j < 4; ++j)
      stage_B(Bk + ((g + j) >> 4) * FRAG, k2, (g + j) & 15,
              kb[(b * CQN + k2) * PP + (g + j) * WW + w]);
  }
  __syncthreads();
  for (int t = wave; t < 36; t += 8) {
    const int th = t / 6, tg = t % 6;
    v8f c = {};
    c = wmma_f16(frag_load(Aq + th * FRAG), frag_load(Bk + tg * FRAG), c);
    const int n = lane & 15, mb = (lane >> 4) * 8;
    const int g = tg * 16 + n;
    const int h0 = th * 16 + mb;
    float* po = attn + (size_t)(b * PP + h0 * WW + w) * 192 + g;
#pragma unroll
    for (int r = 0; r < 8; ++r)
      po[r * (WW * 192)] = ((h0 + r) == g) ? -3.0e38f : c[r];
  }
}

// ---------------------------------------------------------------------------
// Pass 3: per (b,h) row: E_w[w,v] = sum_c q[c,h,w] k[c,h,v] (LDS), then
// 192-way softmax; normalized attention written back (float4 vectorized).
// ---------------------------------------------------------------------------
__global__ void __launch_bounds__(256) row_softmax_kernel(
    const _Float16* __restrict__ qb, const _Float16* __restrict__ kb,
    float* __restrict__ attn) {
  __shared__ __align__(16) _Float16 Aq[6 * FRAG];
  __shared__ __align__(16) _Float16 Bk[6 * FRAG];
  __shared__ __align__(16) float ew[96 * 96];
  const int tid = threadIdx.x;
  const int wave = tid >> 5;
  const int lane = tid & 31;
  const int h = blockIdx.x;
  const int b = blockIdx.y;
#pragma unroll
  for (int cch = 0; cch < 3; ++cch) {
    int e = (tid * 3 + cch) * 4;
    int m = e >> 5, kk = e & 31;        // Aq[w=m][c=kk..kk+3]
    v4h pk;
#pragma unroll
    for (int j = 0; j < 4; ++j)
      pk[j] = qb[(b * CQN + kk + j) * PP + h * WW + m];
    stage_A4(Aq + (m >> 4) * FRAG, m & 15, kk, pk);
    int k2 = e / 96, vv = e % 96;       // Bk[c=k2][v..v+3]: contiguous global read
    v4h kv = *(const v4h*)&kb[(b * CQN + k2) * PP + h * WW + vv];
#pragma unroll
    for (int j = 0; j < 4; ++j)
      stage_B(Bk + ((vv + j) >> 4) * FRAG, k2, (vv + j) & 15, kv[j]);
  }
  __syncthreads();
  for (int t = wave; t < 36; t += 8) {
    const int tw = t / 6, tv = t % 6;
    v8f c = {};
    c = wmma_f16(frag_load(Aq + tw * FRAG), frag_load(Bk + tv * FRAG), c);
    const int n = lane & 15, mb = (lane >> 4) * 8;
#pragma unroll
    for (int r = 0; r < 8; ++r)
      ew[(tw * 16 + mb + r) * 96 + tv * 16 + n] = c[r];
  }
  __syncthreads();
  if (tid < 96) {  // one thread per pixel; float4-vectorized 192-way softmax
    float4* lp = (float4*)(attn + (size_t)(b * PP + h * WW + tid) * 192);
    const float4* ep = (const float4*)(ew + tid * 96);
    float m = -3.0e38f;
    for (int i = 0; i < 24; ++i) {
      float4 a = lp[i];
      m = fmaxf(m, fmaxf(fmaxf(a.x, a.y), fmaxf(a.z, a.w)));
    }
    for (int i = 0; i < 24; ++i) {
      float4 a = ep[i];
      m = fmaxf(m, fmaxf(fmaxf(a.x, a.y), fmaxf(a.z, a.w)));
    }
    float s = 0.f;
    for (int i = 0; i < 24; ++i) {
      float4 a = lp[i];
      s += __expf(a.x - m) + __expf(a.y - m) + __expf(a.z - m) + __expf(a.w - m);
    }
    for (int i = 0; i < 24; ++i) {
      float4 a = ep[i];
      s += __expf(a.x - m) + __expf(a.y - m) + __expf(a.z - m) + __expf(a.w - m);
    }
    const float inv = 1.0f / s;
    for (int i = 0; i < 24; ++i) {
      float4 a = lp[i];
      a.x = __expf(a.x - m) * inv; a.y = __expf(a.y - m) * inv;
      a.z = __expf(a.z - m) * inv; a.w = __expf(a.w - m) * inv;
      lp[i] = a;
    }
    for (int i = 0; i < 24; ++i) {
      float4 a = ep[i];
      a.x = __expf(a.x - m) * inv; a.y = __expf(a.y - m) * inv;
      a.z = __expf(a.z - m) * inv; a.w = __expf(a.w - m) * inv;
      lp[24 + i] = a;
    }
  }
}

// ---------------------------------------------------------------------------
// Pass 4: per (b,w) column: out_h[c,h] = sum_g v[c,g,w] a_h[h,g] -> partial
// (partial buffer stays L2-resident until row_out consumes it)
// ---------------------------------------------------------------------------
__global__ void __launch_bounds__(256) col_out_kernel(
    const _Float16* __restrict__ vb, const float* __restrict__ attn,
    float* __restrict__ outp) {
  __shared__ __align__(16) _Float16 Av[24 * FRAG];  // 8 ct x 3 ko tiles (half of C)
  __shared__ __align__(16) _Float16 Ba[18 * FRAG];  // 3 ko x 6 nt tiles
  const int tid = threadIdx.x;
  const int wave = tid >> 5;
  const int lane = tid & 31;
  const int w = blockIdx.x;
  const int b = blockIdx.y;
  for (int i = 0; i < 9; ++i) {   // Ba[k=g][n=h] = a_h[h][g]; g contiguous in attn
    int e = (tid * 9 + i) * 4;
    int hh = e / 96, g = e % 96;  // 4-aligned g
    float4 av = *(const float4*)&attn[(size_t)(b * PP + hh * WW + w) * 192 + g];
    stage_B4(Ba + ((g >> 5) * 6 + (hh >> 4)) * FRAG, g & 31, hh & 15, cvt4(av));
  }
  for (int half = 0; half < 2; ++half) {
    __syncthreads();
    for (int i = 0; i < 6; ++i) {  // Av[m=c][k=g] = v[c,g,w]; 8-aligned g chunks
      int e = (tid * 6 + i) * 8;
      int cl = e / 96, g = e % 96;
      const _Float16* vp = vb + (size_t)(b * CC + half * 128 + cl) * PP + g * WW + w;
      v8h pk;
#pragma unroll
      for (int j = 0; j < 8; ++j) pk[j] = vp[j * WW];
      stage_A8(Av + ((cl >> 4) * 3 + (g >> 5)) * FRAG, cl & 15, g & 31, pk);
    }
    __syncthreads();
    for (int t = wave; t < 48; t += 8) {
      const int ct = t / 6, nt = t % 6;
      v8f c = {};
#pragma unroll
      for (int ko = 0; ko < 3; ++ko)
        c = wmma_f16(frag_load(Av + (ct * 3 + ko) * FRAG),
                     frag_load(Ba + (ko * 6 + nt) * FRAG), c);
      const int n = lane & 15, mb = (lane >> 4) * 8;
      float* po = outp + (size_t)(b * CC + half * 128 + ct * 16 + mb) * PP
                       + (nt * 16 + n) * WW + w;
#pragma unroll
      for (int r = 0; r < 8; ++r)
        po[r * PP] = c[r];
    }
  }
}

// ---------------------------------------------------------------------------
// Pass 5: per (b,h) row: out_w[c,w] = sum_v v[c,h,v] a_w[w,v];
// final out = gamma * (partial + out_w) + x  (fused residual).
// ---------------------------------------------------------------------------
__global__ void __launch_bounds__(256) row_out_kernel(
    const _Float16* __restrict__ vb, const float* __restrict__ attn,
    const float* __restrict__ xres, const float* __restrict__ gamma,
    float* __restrict__ outp) {
  __shared__ __align__(16) _Float16 Av[24 * FRAG];
  __shared__ __align__(16) _Float16 Ba[18 * FRAG];
  const int tid = threadIdx.x;
  const int wave = tid >> 5;
  const int lane = tid & 31;
  const int h = blockIdx.x;
  const int b = blockIdx.y;
  const float g = gamma[0];
  for (int i = 0; i < 9; ++i) {   // Ba[k=v][n=w] = a_w[w][v]; v contiguous in attn
    int e = (tid * 9 + i) * 4;
    int ww = e / 96, vv = e % 96;
    float4 av = *(const float4*)&attn[(size_t)(b * PP + h * WW + ww) * 192 + 96 + vv];
    stage_B4(Ba + ((vv >> 5) * 6 + (ww >> 4)) * FRAG, vv & 31, ww & 15, cvt4(av));
  }
  for (int half = 0; half < 2; ++half) {
    __syncthreads();
    for (int i = 0; i < 6; ++i) {  // Av[m=c][k=v] = v[c,h,v]; contiguous b128 load
      int e = (tid * 6 + i) * 8;
      int cl = e / 96, vv = e % 96;
      v8h pk = *(const v8h*)&vb[(size_t)(b * CC + half * 128 + cl) * PP + h * WW + vv];
      stage_A8(Av + ((cl >> 4) * 3 + (vv >> 5)) * FRAG, cl & 15, vv & 31, pk);
    }
    __syncthreads();
    for (int t = wave; t < 48; t += 8) {
      const int ct = t / 6, nt = t % 6;
      v8f c = {};
#pragma unroll
      for (int ko = 0; ko < 3; ++ko)
        c = wmma_f16(frag_load(Av + (ct * 3 + ko) * FRAG),
                     frag_load(Ba + (ko * 6 + nt) * FRAG), c);
      const int n = lane & 15, mb = (lane >> 4) * 8;
      const size_t base = (size_t)(b * CC + half * 128 + ct * 16 + mb) * PP
                        + h * WW + nt * 16 + n;
      float* po = outp + base;
      const float* px = xres + base;
#pragma unroll
      for (int r = 0; r < 8; ++r)
        po[r * PP] = g * (po[r * PP] + c[r]) + px[r * PP];
    }
  }
}

extern "C" void kernel_launch(void* const* d_in, const int* in_sizes, int n_in,
                              void* d_out, int out_size, void* d_ws, size_t ws_size,
                              hipStream_t stream) {
  (void)in_sizes; (void)n_in; (void)out_size; (void)ws_size;
  const float* x0  = (const float*)d_in[0];
  const float* x1  = (const float*)d_in[1];
  const float* wq0 = (const float*)d_in[2];
  const float* bq0 = (const float*)d_in[3];
  const float* wk0 = (const float*)d_in[4];
  const float* bk0 = (const float*)d_in[5];
  const float* wv0 = (const float*)d_in[6];
  const float* bv0 = (const float*)d_in[7];
  const float* wq1 = (const float*)d_in[8];
  const float* bq1 = (const float*)d_in[9];
  const float* wk1 = (const float*)d_in[10];
  const float* bk1 = (const float*)d_in[11];
  const float* wv1 = (const float*)d_in[12];
  const float* bv1 = (const float*)d_in[13];
  const float* gam = (const float*)d_in[14];
  float* out0 = (float*)d_out;
  float* out1 = out0 + (size_t)BB * CC * PP;

  // workspace: q0,k0,q1,k1 (f16) + v0,v1 (f16) + attn (f32) ~= 151 MB
  char* ws = (char*)d_ws;
  const size_t qk_bytes = (size_t)BB * CQN * PP * sizeof(_Float16);
  const size_t v_bytes  = (size_t)BB * CC  * PP * sizeof(_Float16);
  _Float16* q0b = (_Float16*)ws; ws += qk_bytes;
  _Float16* k0b = (_Float16*)ws; ws += qk_bytes;
  _Float16* q1b = (_Float16*)ws; ws += qk_bytes;
  _Float16* k1b = (_Float16*)ws; ws += qk_bytes;
  _Float16* v0b = (_Float16*)ws; ws += v_bytes;
  _Float16* v1b = (_Float16*)ws; ws += v_bytes;
  float* attn = (float*)ws;  // B*P*192 f32, reused across both attends (stream-ordered)

  dim3 projGrid(PP / 128, 320 / 16, BB);
  qkv_proj_kernel<<<projGrid, 256, 0, stream>>>(x0, wq0, bq0, wk0, bk0, wv0, bv0, q0b, k0b, v0b);
  qkv_proj_kernel<<<projGrid, 256, 0, stream>>>(x1, wq1, bq1, wk1, bk1, wv1, bv1, q1b, k1b, v1b);

  dim3 colGrid(WW, BB), rowGrid(HH, BB);
  // attend 0: q1, k0, v0, residual x0 -> out0
  col_logits_kernel<<<colGrid, 256, 0, stream>>>(q1b, k0b, attn);
  row_softmax_kernel<<<rowGrid, 256, 0, stream>>>(q1b, k0b, attn);
  col_out_kernel<<<colGrid, 256, 0, stream>>>(v0b, attn, out0);
  row_out_kernel<<<rowGrid, 256, 0, stream>>>(v0b, attn, x0, gam, out0);
  // attend 1: q0, k1, v1, residual x1 -> out1
  col_logits_kernel<<<colGrid, 256, 0, stream>>>(q0b, k1b, attn);
  row_softmax_kernel<<<rowGrid, 256, 0, stream>>>(q0b, k1b, attn);
  col_out_kernel<<<colGrid, 256, 0, stream>>>(v1b, attn, out1);
  row_out_kernel<<<rowGrid, 256, 0, stream>>>(v1b, attn, x1, gam, out1);
}
